// DLRM_53721450938756
// MI455X (gfx1250) — matrix operand order repaired and checked by reading, not compiled, not run on hardware
//
#include <hip/hip_runtime.h>
#include <hip/hip_bf16.h>
#include <math.h>

// ---------------- types ----------------
typedef __attribute__((ext_vector_type(16))) _Float16 v16h;
typedef __attribute__((ext_vector_type(8)))  _Float16 v8h;
typedef __attribute__((ext_vector_type(8)))  float    v8f;
typedef __attribute__((ext_vector_type(4)))  float    v4f;

union HV { v16h v; v8h h[2]; };

#define BATCH   16384
#define N_CONT  13
#define N_CAT   26
#define VOCAB   50000
#define EMB     64
#define BOT_H   512
#define TOP_H   1024
#define C_FEAT  27            // N_CAT + 1
#define INTER   378           // 27*28/2
#define TOP_IN  442           // 64 + 378
#define TOP_INP 448           // padded to multiple of 32

#define WMMA_F16(A, B, Cacc) \
    __builtin_amdgcn_wmma_f32_16x16x32_f16(false, (A), false, (B), (short)0, (Cacc), false, false)

// ============================================================
// f16 GEMM:  C[M,N] = act( A[M,K] * Bt[N,K]^T + bias )
// A row-major [M, lda], Bt row-major [N, ldb] (B pre-transposed).
// Block = 256 threads = 8 waves; each wave computes a 64(M) x 32(N)
// output block (4x2 WMMA tiles): 12 b128 loads -> 8 WMMAs per K-step
// (1.5 loads/WMMA).  Block tile = 512(M) x 32(N); waves stacked along
// M so all 8 waves share the same B strip (L1 hits).
// grid = (N/32, M/512).  fp32 accumulation.
// ============================================================
__global__ __launch_bounds__(256)
void dlrm_gemm_f16(const _Float16* __restrict__ A, int lda,
                   const _Float16* __restrict__ Bt, int ldb,
                   const float* __restrict__ bias,
                   _Float16* __restrict__ C, int ldc,
                   int K, int act)
{
    const int lane = threadIdx.x & 31;
    const int wave = threadIdx.x >> 5;
    const int hi   = lane >> 4;          // 0: lanes 0-15, 1: lanes 16-31
    const int l15  = lane & 15;

    const int n0 = blockIdx.x * 32;
    const int m0 = blockIdx.y * 512 + wave * 64;

    const _Float16* arow[4];
#pragma unroll
    for (int i = 0; i < 4; ++i)
        arow[i] = A + (size_t)(m0 + 16 * i + l15) * lda;
    const _Float16* brow[2];
#pragma unroll
    for (int j = 0; j < 2; ++j)
        brow[j] = Bt + (size_t)(n0 + 16 * j + l15) * ldb;

    v8f acc[8] = {};
    for (int k0 = 0; k0 < K; k0 += 32) {
        HV a[4], b[2];
        // A fragment: lane<16 holds K {k0..k0+7, k0+16..k0+23}; lane>=16: +8
        const int ka = k0 + (hi ? 8 : 0);
#pragma unroll
        for (int i = 0; i < 4; ++i) {
            a[i].h[0] = *(const v8h*)(arow[i] + ka);
            a[i].h[1] = *(const v8h*)(arow[i] + ka + 16);
        }
        // B fragment: lane<16 holds K {k0..k0+7, k0+16..k0+23}; lane>=16: +8
        const int kb = k0 + (hi ? 8 : 0);
#pragma unroll
        for (int j = 0; j < 2; ++j) {
            b[j].h[0] = *(const v8h*)(brow[j] + kb);
            b[j].h[1] = *(const v8h*)(brow[j] + kb + 16);
        }
#pragma unroll
        for (int i = 0; i < 4; ++i)
#pragma unroll
            for (int j = 0; j < 2; ++j)
                acc[i * 2 + j] = WMMA_F16(a[i].v, b[j].v, acc[i * 2 + j]);
    }

    const float bv0 = bias ? bias[n0 + l15]      : 0.0f;
    const float bv1 = bias ? bias[n0 + 16 + l15] : 0.0f;
#pragma unroll
    for (int i = 0; i < 4; ++i) {
#pragma unroll
        for (int j = 0; j < 2; ++j) {
            const int   n  = n0 + 16 * j + l15;
            const float bv = j ? bv1 : bv0;
#pragma unroll
            for (int r = 0; r < 8; ++r) {
                const int m = m0 + 16 * i + r + (hi ? 8 : 0);
                float v = acc[i * 2 + j][r] + bv;
                if      (act == 1) v = v > 0.0f ? v : 0.0f;           // relu
                else if (act == 2) v = 1.0f / (1.0f + __expf(-v));    // sigmoid
                C[(size_t)m * ldc + n] = (_Float16)v;
            }
        }
    }
}

// ============================================================
// Interaction: per sample, features F = [26 gathered emb rows ; bottom] (27x64).
// Gram = F * F^T via WMMA (pad M,N to 32); write upper triangle (378 vals)
// into final[:, 64..441]; final[:, 0..63] already holds bottom (from GEMM3);
// final[:, 442..447] zero-padded here.
// One wave per sample, 8 samples per block.
// ============================================================
__device__ inline v8h dlrm_feat8(const float* __restrict__ emb,
                                 const int*   __restrict__ cat,
                                 const _Float16* __restrict__ frow,
                                 int b, int c, int k)
{
    v8h r = {};
    if (c < N_CAT) {
        const float* p = emb + ((size_t)c * VOCAB + (size_t)cat[b * N_CAT + c]) * EMB + k;
        v4f lo = *(const v4f*)p;
        v4f hh = *(const v4f*)(p + 4);
#pragma unroll
        for (int i = 0; i < 4; ++i) { r[i] = (_Float16)lo[i]; r[i + 4] = (_Float16)hh[i]; }
    } else if (c == N_CAT) {
        r = *(const v8h*)(frow + k);   // bottom output lives in final[:,0..63]
    }
    return r;
}

__global__ __launch_bounds__(256)
void dlrm_interact(const float* __restrict__ emb,
                   const int*   __restrict__ cat,
                   _Float16*    __restrict__ finalh)
{
    const int lane = threadIdx.x & 31;
    const int wave = threadIdx.x >> 5;
    const int b    = blockIdx.x * 8 + wave;
    const int hi   = lane >> 4;
    const int l15  = lane & 15;

    _Float16* frow = finalh + (size_t)b * TOP_INP;

    // zero the K-padding columns 442..447 for the following GEMM
    if (lane < TOP_INP - TOP_IN) frow[TOP_IN + lane] = (_Float16)0.0f;

    v8f c00 = {}, c01 = {}, c11 = {};
#pragma unroll
    for (int ks = 0; ks < 2; ++ks) {        // K = 64 = 2 x 32
        const int ka = ks * 32 + (hi ? 8 : 0);
        const int kb = ks * 32 + (hi ? 8 : 0);
        HV a0, a1, b0, b1;
        a0.h[0] = dlrm_feat8(emb, cat, frow, b, l15,      ka);
        a0.h[1] = dlrm_feat8(emb, cat, frow, b, l15,      ka + 16);
        a1.h[0] = dlrm_feat8(emb, cat, frow, b, 16 + l15, ka);
        a1.h[1] = dlrm_feat8(emb, cat, frow, b, 16 + l15, ka + 16);
        b0.h[0] = dlrm_feat8(emb, cat, frow, b, l15,      kb);
        b0.h[1] = dlrm_feat8(emb, cat, frow, b, l15,      kb + 16);
        b1.h[0] = dlrm_feat8(emb, cat, frow, b, 16 + l15, kb);
        b1.h[1] = dlrm_feat8(emb, cat, frow, b, 16 + l15, kb + 16);
        c00 = WMMA_F16(a0.v, b0.v, c00);
        c01 = WMMA_F16(a0.v, b1.v, c01);
        c11 = WMMA_F16(a1.v, b1.v, c11);
    }

    // upper triangle (i<=j), row-major pair order -> final[:, 64 + pos]
    const v8f tiles[3] = { c00, c01, c11 };
    const int mts[3] = {0, 0, 1}, nts[3] = {0, 1, 1};
#pragma unroll
    for (int t = 0; t < 3; ++t) {
        const int j = nts[t] * 16 + l15;
#pragma unroll
        for (int r = 0; r < 8; ++r) {
            const int i = mts[t] * 16 + r + (hi ? 8 : 0);
            if (i <= j && j < C_FEAT) {
                const int pos = i * C_FEAT - (i * (i - 1)) / 2 + (j - i);
                frow[EMB + pos] = (_Float16)tiles[t][r];
            }
        }
    }
}

// ============================================================
// Weight prep: transpose + fp32->f16 + zero-pad K dimension.
// dst[n, k] = (k < K) ? src[k, n] : 0     dst is [N, Kpad]
// ============================================================
__global__ void dlrm_wprep(const float* __restrict__ src, _Float16* __restrict__ dst,
                           int K, int N, int Kpad)
{
    int idx = blockIdx.x * 256 + threadIdx.x;
    if (idx >= N * Kpad) return;
    int n = idx / Kpad, k = idx - n * Kpad;
    dst[idx] = (k < K) ? (_Float16)src[(size_t)k * N + n] : (_Float16)0.0f;
}

// continuous [B,13] fp32 -> [B,32] f16 zero-padded
__global__ void dlrm_contprep(const float* __restrict__ src, _Float16* __restrict__ dst)
{
    int idx = blockIdx.x * 256 + threadIdx.x;
    if (idx >= BATCH * 32) return;
    int b = idx >> 5, k = idx & 31;
    dst[idx] = (k < N_CONT) ? (_Float16)src[b * N_CONT + k] : (_Float16)0.0f;
}

// ============================================================
// Final layer: out[b] = dot(t2[b,:], w_to) + b_to   (N=1, wave-per-row
// fp32 reduction instead of a degenerate GEMM)
// ============================================================
__global__ __launch_bounds__(256)
void dlrm_topout(const _Float16* __restrict__ t2, const float* __restrict__ wto,
                 const float* __restrict__ bto, float* __restrict__ out)
{
    const int lane = threadIdx.x & 31;
    const int wave = threadIdx.x >> 5;
    const int b = blockIdx.x * 8 + wave;
    const _Float16* row = t2 + (size_t)b * TOP_H;
    float acc = 0.0f;
    for (int k = lane; k < TOP_H; k += 32) acc += (float)row[k] * wto[k];
#pragma unroll
    for (int off = 16; off; off >>= 1) acc += __shfl_down(acc, off, 32);
    if (lane == 0) out[b] = acc + bto[0];
}

// ============================================================
extern "C" void kernel_launch(void* const* d_in, const int* in_sizes, int n_in,
                              void* d_out, int out_size, void* d_ws, size_t ws_size,
                              hipStream_t stream)
{
    const float* cont  = (const float*)d_in[0];
    const int*   cat   = (const int*)  d_in[1];
    const float* emb   = (const float*)d_in[2];
    const float* w_b1  = (const float*)d_in[3];
    const float* b_b1  = (const float*)d_in[4];
    const float* w_bh  = (const float*)d_in[5];
    const float* b_bh  = (const float*)d_in[6];
    const float* w_bo  = (const float*)d_in[7];
    const float* b_bo  = (const float*)d_in[8];
    const float* w_t1  = (const float*)d_in[9];
    const float* b_t1  = (const float*)d_in[10];
    const float* w_th  = (const float*)d_in[11];
    const float* b_th  = (const float*)d_in[12];
    const float* w_to  = (const float*)d_in[13];
    const float* b_to  = (const float*)d_in[14];
    float* out = (float*)d_out;

    char* ws = (char*)d_ws;
    size_t off = 0;
    auto alloc = [&](size_t bytes) -> void* {
        void* p = ws + off;
        off += (bytes + 255) & ~(size_t)255;
        return p;
    };

    _Float16* conth = (_Float16*)alloc((size_t)BATCH * 32 * 2);
    _Float16* wtb1  = (_Float16*)alloc((size_t)BOT_H * 32 * 2);
    _Float16* wtbh  = (_Float16*)alloc((size_t)BOT_H * BOT_H * 2);
    _Float16* wtbo  = (_Float16*)alloc((size_t)EMB * BOT_H * 2);
    _Float16* wtt1  = (_Float16*)alloc((size_t)TOP_H * TOP_INP * 2);
    _Float16* wtth  = (_Float16*)alloc((size_t)TOP_H * TOP_H * 2);
    _Float16* h1    = (_Float16*)alloc((size_t)BATCH * BOT_H * 2);
    _Float16* h2    = (_Float16*)alloc((size_t)BATCH * BOT_H * 2);
    _Float16* fin   = (_Float16*)alloc((size_t)BATCH * TOP_INP * 2);
    _Float16* t1    = (_Float16*)alloc((size_t)BATCH * TOP_H * 2);
    _Float16* t2    = (_Float16*)alloc((size_t)BATCH * TOP_H * 2);
    (void)ws_size; (void)in_sizes; (void)n_in; (void)out_size;

    auto cdiv = [](int a, int b) { return (a + b - 1) / b; };

    // ---- weight / input prep (cheap, once per call) ----
    dlrm_contprep<<<cdiv(BATCH * 32, 256), 256, 0, stream>>>(cont, conth);
    dlrm_wprep<<<cdiv(BOT_H * 32, 256),        256, 0, stream>>>(w_b1, wtb1, N_CONT, BOT_H, 32);
    dlrm_wprep<<<cdiv(BOT_H * BOT_H, 256),     256, 0, stream>>>(w_bh, wtbh, BOT_H, BOT_H, BOT_H);
    dlrm_wprep<<<cdiv(EMB * BOT_H, 256),       256, 0, stream>>>(w_bo, wtbo, BOT_H, EMB, BOT_H);
    dlrm_wprep<<<cdiv(TOP_H * TOP_INP, 256),   256, 0, stream>>>(w_t1, wtt1, TOP_IN, TOP_H, TOP_INP);
    dlrm_wprep<<<cdiv(TOP_H * TOP_H, 256),     256, 0, stream>>>(w_th, wtth, TOP_H, TOP_H, TOP_H);

    // ---- bottom MLP ----
    dlrm_gemm_f16<<<dim3(BOT_H / 32, BATCH / 512), 256, 0, stream>>>(
        conth, 32, wtb1, 32, b_b1, h1, BOT_H, 32, /*relu*/1);
    dlrm_gemm_f16<<<dim3(BOT_H / 32, BATCH / 512), 256, 0, stream>>>(
        h1, BOT_H, wtbh, BOT_H, b_bh, h2, BOT_H, BOT_H, /*relu*/1);
    // bottom output (sigmoid) goes straight into final[:, 0..63]
    dlrm_gemm_f16<<<dim3(EMB / 32, BATCH / 512), 256, 0, stream>>>(
        h2, BOT_H, wtbo, BOT_H, b_bo, fin, TOP_INP, BOT_H, /*sigmoid*/2);

    // ---- feature interaction (emb gather + Gram upper triangle) ----
    dlrm_interact<<<BATCH / 8, 256, 0, stream>>>(emb, cat, fin);

    // ---- top MLP ----
    dlrm_gemm_f16<<<dim3(TOP_H / 32, BATCH / 512), 256, 0, stream>>>(
        fin, TOP_INP, wtt1, TOP_INP, b_t1, t1, TOP_H, TOP_INP, /*relu*/1);
    dlrm_gemm_f16<<<dim3(TOP_H / 32, BATCH / 512), 256, 0, stream>>>(
        t1, TOP_H, wtth, TOP_H, b_th, t2, TOP_H, TOP_H, /*relu*/1);
    dlrm_topout<<<BATCH / 8, 256, 0, stream>>>(t2, w_to, b_to, out);
}
